// TransformerDecoder_14224931684504
// MI455X (gfx1250) — compile-verified, matrix operand order checked by reading, not run on hardware
//
#include <hip/hip_runtime.h>
#include <hip/hip_bf16.h>

// ---------------------------------------------------------------------------
// Transformer decoder forward on MI455X (gfx1250), wave32 + WMMA bf16.
// All GEMMs run through v_wmma_f32_16x16x32_bf16 (fp32 accumulate).
// ---------------------------------------------------------------------------

#define TB 8
#define TS 1024
#define TH 1024
#define TV 512
#define TL 6
#define NEG_INF (-10000000.0f)

typedef __attribute__((ext_vector_type(16))) __bf16 bf16x16;
typedef __attribute__((ext_vector_type(8)))  float  f32x8;

union Frag { bf16x16 v; unsigned int u[8]; };

static __device__ __forceinline__ unsigned short f2bf(float f) {
  unsigned int u = __float_as_uint(f);
  u += 0x7FFFu + ((u >> 16) & 1u);          // round-to-nearest-even
  return (unsigned short)(u >> 16);
}

// ---------------------------------------------------------------------------
// Embedding gather: x_f32 and x_bf16 copies.
// grid = B*S blocks, 256 threads, 4 elems/thread (H=1024).
// ---------------------------------------------------------------------------
__global__ __launch_bounds__(256) void embed_kernel(
    const int* __restrict__ inp, const float* __restrict__ emb,
    float* __restrict__ xf, unsigned short* __restrict__ xb) {
  int tok = blockIdx.x;
  int idx = inp[tok];
  const float* e = emb + (size_t)idx * TH;
  size_t o = (size_t)tok * TH;
#pragma unroll
  for (int i = 0; i < 4; ++i) {
    int h = threadIdx.x + i * 256;
    float v = e[h];
    xf[o + h] = v;
    xb[o + h] = f2bf(v);
  }
}

// ---------------------------------------------------------------------------
// Generic WMMA GEMM:  C = epilogue( A_bf16[M,K] @ B[K,N] * scale )
//   BT      : float (weights, converted to bf16 on LDS store) or ushort (bf16)
//   BTRANS  : B stored [N][K] row-major (used for logits: B = K-matrix rows)
//   MASK    : add NEG_INF where n > m (causal, logits laid out [q][k])
//   RELU    : relu after bias
//   RES2    : add res1[m,n] + res2[m,n] after activation
//   OUTF/OUTB : write fp32 / bf16 outputs
// Block tile 128(M) x 64(N); 8 waves of 32 lanes; wave tile 32x32 (2x2 WMMA).
// K loop in steps of 32 with double-buffered LDS.
// ---------------------------------------------------------------------------
template <typename BT, bool BTRANS, bool MASK, bool RELU, bool RES2,
          bool OUTF, bool OUTB>
__global__ __launch_bounds__(256) void wmma_gemm(
    const unsigned short* __restrict__ A, int lda, long long aBatch,
    const BT* __restrict__ Bm, int ldb, long long bBatch,
    const float* __restrict__ bias,
    const float* __restrict__ res1, const float* __restrict__ res2,
    float scale,
    float* __restrict__ outF, unsigned short* __restrict__ outB,
    int ldo, long long oBatch, int K) {
  // LDS: A tile 128x32 bf16, B tile stored transposed [n][k] 64x32 bf16.
  // Pitch 40 halfs (80B) -> 16B-aligned rows, conflict-spread fragment reads.
  __shared__ __align__(16) unsigned short As[2][128 * 40];
  __shared__ __align__(16) unsigned short Bs[2][64 * 40];

  const int mt = blockIdx.y * 128;
  const int nt = blockIdx.x * 64;
  const size_t aOff = (size_t)blockIdx.z * (size_t)aBatch;
  const size_t bOff = (size_t)blockIdx.z * (size_t)bBatch;
  const size_t oOff = (size_t)blockIdx.z * (size_t)oBatch;

  const int lane = threadIdx.x & 31;   // wave32
  const int wave = threadIdx.x >> 5;   // 0..7
  const int wm = wave & 3;             // M sub-tile (32 rows each)
  const int wn = wave >> 2;            // N sub-tile (32 cols each)
  const int half = lane >> 4;          // lane group per ISA 7.12.2
  const int l16 = lane & 15;

  auto loadTiles = [&](int buf, int kt) {
    // ---- A tile: 128 rows x 32 halfs; 2 threads/row, 2x b128 each ----
    {
      int row = threadIdx.x >> 1, hh = threadIdx.x & 1;
      const uint4* ap =
          (const uint4*)(A + aOff + (size_t)(mt + row) * lda + kt + hh * 16);
      uint4* dst = (uint4*)&As[buf][row * 40 + hh * 16];
      dst[0] = ap[0];
      dst[1] = ap[1];
    }
    // ---- B tile -> LDS [n][k] ----
    if constexpr (BTRANS) {
      // B is bf16 [N][K]: direct row copy (already [n][k]).
      if (threadIdx.x < 128) {
        int n = threadIdx.x >> 1, hh = threadIdx.x & 1;
        const uint4* bp =
            (const uint4*)(Bm + bOff + (size_t)(nt + n) * ldb + kt + hh * 16);
        uint4* dst = (uint4*)&Bs[buf][n * 40 + hh * 16];
        dst[0] = bp[0];
        dst[1] = bp[1];
      }
    } else if constexpr (sizeof(BT) == 4) {
      // B is f32 [K][N] (weights): load row chunk, convert + transpose.
      int k = threadIdx.x >> 3;
      int n0 = (threadIdx.x & 7) * 8;
      const float4* bp =
          (const float4*)(Bm + bOff + (size_t)(kt + k) * ldb + nt + n0);
      float4 q0 = bp[0], q1 = bp[1];
      float vv[8] = {q0.x, q0.y, q0.z, q0.w, q1.x, q1.y, q1.z, q1.w};
#pragma unroll
      for (int j = 0; j < 8; ++j) Bs[buf][(n0 + j) * 40 + k] = f2bf(vv[j]);
    } else {
      // B is bf16 [K][N] (V matrix): load row chunk, transpose scatter.
      int k = threadIdx.x >> 3;
      int n0 = (threadIdx.x & 7) * 8;
      uint4 pv = *(const uint4*)(Bm + bOff + (size_t)(kt + k) * ldb + nt + n0);
      const unsigned short* hv = (const unsigned short*)&pv;
#pragma unroll
      for (int j = 0; j < 8; ++j) Bs[buf][(n0 + j) * 40 + k] = hv[j];
    }
  };

  f32x8 acc[2][2] = {};
  const int nk = K >> 5;

  loadTiles(0, 0);
  __syncthreads();

  for (int kb = 0; kb < nk; ++kb) {
    const int cur = kb & 1;
    if (kb + 1 < nk) loadTiles(cur ^ 1, (kb + 1) << 5);

    Frag af[2], bfm[2];
#pragma unroll
    for (int mi = 0; mi < 2; ++mi) {
      int rbase = (wm * 32 + mi * 16 + l16) * 40;
#pragma unroll
      for (int v = 0; v < 8; ++v) {
        int k0 = ((v >> 2) << 4) + half * 8 + ((v & 3) << 1);
        af[mi].u[v] = *(const unsigned int*)&As[cur][rbase + k0];
      }
    }
#pragma unroll
    for (int ni = 0; ni < 2; ++ni) {
      int cbase = (wn * 32 + ni * 16 + l16) * 40;
#pragma unroll
      for (int v = 0; v < 8; ++v) {
        int k0 = ((v >> 2) << 4) + half * 8 + ((v & 3) << 1);
        bfm[ni].u[v] = *(const unsigned int*)&Bs[cur][cbase + k0];
      }
    }
#pragma unroll
    for (int mi = 0; mi < 2; ++mi)
#pragma unroll
      for (int ni = 0; ni < 2; ++ni)
        acc[mi][ni] = __builtin_amdgcn_wmma_f32_16x16x32_bf16(
            false, af[mi].v, false, bfm[ni].v, (short)0, acc[mi][ni], false,
            false);

    __syncthreads();
  }

  // Epilogue. C layout (ISA 7.12.2): m = base + 8*half + r, n = base + l16.
#pragma unroll
  for (int mi = 0; mi < 2; ++mi) {
#pragma unroll
    for (int ni = 0; ni < 2; ++ni) {
#pragma unroll
      for (int r = 0; r < 8; ++r) {
        int m = mt + wm * 32 + mi * 16 + half * 8 + r;
        int n = nt + wn * 32 + ni * 16 + l16;
        float v = acc[mi][ni][r] * scale;
        if constexpr (MASK) {
          if (n > m) v += NEG_INF;  // logits [q][k]: mask keys k > q
        }
        if (bias) v += bias[n];
        if constexpr (RELU) v = fmaxf(v, 0.0f);
        size_t o = oOff + (size_t)m * ldo + n;
        if constexpr (RES2) v += res1[o] + res2[o];
        if constexpr (OUTF) outF[o] = v;
        if constexpr (OUTB) outB[o] = f2bf(v);
      }
    }
  }
}

// ---------------------------------------------------------------------------
// Softmax over k for logits [b][q][k] (contiguous reduce axis).
// Writes fp32 attn to d_out in reference layout [b][k][q], and a bf16
// transposed copy attnT [b][q][k] used as WMMA A-operand of attn^T @ V.
// grid = B*S blocks, 256 threads, 4 k/thread.
// ---------------------------------------------------------------------------
__global__ __launch_bounds__(256) void softmax_kernel(
    const float* __restrict__ logits, float* __restrict__ attnL,
    unsigned short* __restrict__ attnT) {
  __shared__ float red[256];
  const int t = threadIdx.x;
  const int bq = blockIdx.x;
  const int b = bq / TS;
  const int q = bq % TS;
  const float* row = logits + (size_t)bq * TS;

  float v[4];
  float mx = -3.4e38f;
#pragma unroll
  for (int i = 0; i < 4; ++i) {
    v[i] = row[t + i * 256];
    mx = fmaxf(mx, v[i]);
  }
  red[t] = mx;
  __syncthreads();
  for (int s = 128; s > 0; s >>= 1) {
    if (t < s) red[t] = fmaxf(red[t], red[t + s]);
    __syncthreads();
  }
  mx = red[0];
  __syncthreads();

  float e[4], sum = 0.0f;
#pragma unroll
  for (int i = 0; i < 4; ++i) {
    e[i] = __expf(v[i] - mx);
    sum += e[i];
  }
  red[t] = sum;
  __syncthreads();
  for (int s = 128; s > 0; s >>= 1) {
    if (t < s) red[t] += red[t + s];
    __syncthreads();
  }
  const float inv = 1.0f / red[0];

  float* ab = attnL + (size_t)b * TS * TS;  // [k][q] within batch
  unsigned short* at = attnT + (size_t)bq * TS;
#pragma unroll
  for (int i = 0; i < 4; ++i) {
    int k = t + i * 256;
    float p = e[i] * inv;
    ab[(size_t)k * TS + q] = p;  // reference attn layout [b, k, q]
    at[k] = f2bf(p);             // [b, q, k] bf16 for ctx GEMM
  }
}

// ---------------------------------------------------------------------------
// Host orchestration.
// ---------------------------------------------------------------------------
extern "C" void kernel_launch(void* const* d_in, const int* in_sizes, int n_in,
                              void* d_out, int out_size, void* d_ws,
                              size_t ws_size, hipStream_t stream) {
  (void)in_sizes; (void)n_in; (void)out_size; (void)ws_size;

  const int* inputs      = (const int*)d_in[0];
  const float* embedding = (const float*)d_in[1];
  const float* Wq = (const float*)d_in[2];
  const float* bq = (const float*)d_in[3];
  const float* Wk = (const float*)d_in[4];
  const float* bk = (const float*)d_in[5];
  const float* Wv = (const float*)d_in[6];
  const float* bv = (const float*)d_in[7];
  const float* Wm = (const float*)d_in[8];
  const float* bm = (const float*)d_in[9];
  const float* Wout = (const float*)d_in[10];
  const float* bout = (const float*)d_in[11];

  float* out = (float*)d_out;                       // [B,S,V] fp32
  float* attns = out + (size_t)TB * TS * TV;        // [L,B,S,S] fp32

  // Scratch layout (all fully written before read each call).
  char* ws = (char*)d_ws;
  size_t off = 0;
  auto alloc = [&](size_t bytes) -> void* {
    void* p = ws + off;
    off = (off + bytes + 255) & ~(size_t)255;
    return p;
  };
  const size_t BS = (size_t)TB * TS;
  unsigned short* x_bf   = (unsigned short*)alloc(BS * TH * 2);
  float*          x_f    = (float*)alloc(BS * TH * 4);
  unsigned short* q_bf   = (unsigned short*)alloc(BS * TH * 2);
  unsigned short* k_bf   = (unsigned short*)alloc(BS * TH * 2);
  unsigned short* v_bf   = (unsigned short*)alloc(BS * TH * 2);
  unsigned short* attnT  = (unsigned short*)alloc((size_t)TB * TS * TS * 2);
  float*          ctx_f  = (float*)alloc(BS * TH * 4);
  unsigned short* ctx_bf = (unsigned short*)alloc(BS * TH * 2);
  float*          logits = (float*)alloc((size_t)TB * TS * TS * 4);

  const dim3 blk(256);
  const float scale = 0.03125f;  // rsqrt(1024)

  // 1) Embedding gather.
  embed_kernel<<<dim3(BS), blk, 0, stream>>>(inputs, embedding, x_f, x_bf);

  const dim3 gDense(TH / 64, BS / 128, 1);   // [8192 x 1024] tiles
  const dim3 gAttn(TS / 64, TS / 128, TB);   // per-batch [1024 x 1024]
  const long long SH = (long long)TS * TH;
  const long long SS = (long long)TS * TS;

  for (int l = 0; l < TL; ++l) {
    const float* Wql = Wq + (size_t)l * TH * TH;
    const float* Wkl = Wk + (size_t)l * TH * TH;
    const float* Wvl = Wv + (size_t)l * TH * TH;
    const float* Wml = Wm + (size_t)l * TH * TH;
    const float* bql = bq + (size_t)l * TH;
    const float* bkl = bk + (size_t)l * TH;
    const float* bvl = bv + (size_t)l * TH;
    const float* bml = bm + (size_t)l * TH;

    // q/k/v = x @ W + b  -> bf16
    wmma_gemm<float, false, false, false, false, false, true>
        <<<gDense, blk, 0, stream>>>(x_bf, TH, 0, Wql, TH, 0, bql, nullptr,
                                     nullptr, 1.0f, nullptr, q_bf, TH, 0, TH);
    wmma_gemm<float, false, false, false, false, false, true>
        <<<gDense, blk, 0, stream>>>(x_bf, TH, 0, Wkl, TH, 0, bkl, nullptr,
                                     nullptr, 1.0f, nullptr, k_bf, TH, 0, TH);
    wmma_gemm<float, false, false, false, false, false, true>
        <<<gDense, blk, 0, stream>>>(x_bf, TH, 0, Wvl, TH, 0, bvl, nullptr,
                                     nullptr, 1.0f, nullptr, v_bf, TH, 0, TH);

    // logits[b][q][k] = scale * q.k^T + causal mask   (B operand = K rows)
    wmma_gemm<unsigned short, true, true, false, false, true, false>
        <<<gAttn, blk, 0, stream>>>(q_bf, TH, SH, k_bf, TH, SH, nullptr,
                                    nullptr, nullptr, scale, logits, nullptr,
                                    TS, SS, TH);

    // softmax over k -> attn (d_out, [b,k,q]) + attnT bf16 ([b,q,k])
    softmax_kernel<<<dim3(BS), blk, 0, stream>>>(
        logits, attns + (size_t)l * TB * TS * TS, attnT);

    // ctx[b][q][h] = attn^T @ V
    wmma_gemm<unsigned short, false, false, false, false, true, true>
        <<<gAttn, blk, 0, stream>>>(attnT, TS, SS, v_bf, TH, SH, nullptr,
                                    nullptr, nullptr, 1.0f, ctx_f, ctx_bf, TH,
                                    SH, TS);

    // x = x + ctx + relu(ctx @ Wm + bm)  -> fp32 (in place) + bf16
    wmma_gemm<float, false, false, true, true, true, true>
        <<<gDense, blk, 0, stream>>>(ctx_bf, TH, 0, Wml, TH, 0, bml, x_f,
                                     ctx_f, 1.0f, x_f, x_bf, TH, 0, TH);
  }

  // out = x @ Wout + bout
  wmma_gemm<float, false, false, false, false, true, false>
      <<<dim3(TV / 64, BS / 128, 1), blk, 0, stream>>>(
          x_bf, TH, 0, Wout, TV, 0, bout, nullptr, nullptr, 1.0f, out, nullptr,
          TV, 0, TH);
}
